// AutoCorrelation_45140106281304
// MI455X (gfx1250) — matrix operand order repaired and checked by reading, not compiled, not run on hardware
//
#include <hip/hip_runtime.h>
#include <hip/hip_bf16.h>

typedef __attribute__((ext_vector_type(16))) _Float16 v16h;
typedef __attribute__((ext_vector_type(8)))  _Float16 v8h;
typedef __attribute__((ext_vector_type(8)))  float    v8f;

#define TDIM 4096
#define DDIM 512
#define BDIM 8
#define LDA_P 40   // LDS row stride in f16 (32 + 8 pad)

// ---------------- LDS staging helpers ----------------
// Tile is 128 rows x 32 cols (f16) in LDS, row stride LDA_P.
// 256 threads: 2 threads per row, each moves 16 contiguous elements.

__device__ inline void stage_f32(_Float16* lds, const float* __restrict__ src,
                                 int ldsrc, int tid) {
  const int row = tid >> 1;
  const int col = (tid & 1) * 16;
  const float* p = src + row * ldsrc + col;
  v8h h0, h1;
#pragma unroll
  for (int e = 0; e < 8; ++e) {
    h0[e] = (_Float16)p[e];
    h1[e] = (_Float16)p[e + 8];
  }
  *(v8h*)(lds + row * LDA_P + col)     = h0;
  *(v8h*)(lds + row * LDA_P + col + 8) = h1;
}

// f16 source: use the CDNA5 async global->LDS DMA path (no VGPR bounce).
// Each thread issues two GLOBAL_LOAD_ASYNC_TO_LDS_B128 (32 bytes total).
// Tracked by ASYNCcnt; caller must s_wait_asynccnt before the barrier.
__device__ inline void stage_f16_async(_Float16* lds, const _Float16* __restrict__ src,
                                       int ldsrc, int tid) {
  const int row = tid >> 1;
  const int col = (tid & 1) * 16;
  const _Float16* p = src + row * ldsrc + col;
  // addrspace(3) generic address carries the LDS byte offset in addr[31:0]
  const unsigned lds_off = (unsigned)(size_t)(lds + row * LDA_P + col);
  asm volatile(
      "global_load_async_to_lds_b128 %0, %1, off\n\t"
      "global_load_async_to_lds_b128 %2, %3, off"
      :
      : "v"(lds_off), "v"(p), "v"(lds_off + 16u), "v"(p + 8)
      : "memory");
}

__device__ inline void wait_async_lds() {
  asm volatile("s_wait_asynccnt 0" ::: "memory");
}

// ---------------- fragment loaders (ISA 7.12.2 layouts) ----------------
// A (16x32 f16): lane L row = L&15; elems 0..7 -> K = lh*8 + e,
//                elems 8..15 -> K = 16 + lh*8 + (e-8),  lh = L>>4.
__device__ inline v16h load_fragA(const _Float16* lds, int row, int lh) {
  union { v16h v; v8h h[2]; } u;
  u.h[0] = *(const v8h*)(lds + row * LDA_P + lh * 8);
  u.h[1] = *(const v8h*)(lds + row * LDA_P + 16 + lh * 8);
  return u.v;
}
// B (32x16 f16): lane L col = L&15; elems 0..7 -> K = lh*8 + e, elems 8..15 -> K = 16 + lh*8 + (e-8).
// Stored row-major-by-N in LDS (operand is W^T / K^T so rows read directly).
__device__ inline v16h load_fragB(const _Float16* lds, int row, int lh) {
  union { v16h v; v8h h[2]; } u;
  u.h[0] = *(const v8h*)(lds + row * LDA_P + lh * 8);
  u.h[1] = *(const v8h*)(lds + row * LDA_P + 16 + lh * 8);
  return u.v;
}

// ---------------- projection GEMM: out = X @ W^T + bias ----------------
// X: (32768, 512) f32 row-major, W: (512, 512) f32 row-major.
// Block computes 128x128; 8 waves in 2(M) x 4(N) grid; wave tile 64x32.
template <typename OutT>
__global__ __launch_bounds__(256) void proj_gemm(const float* __restrict__ X,
                                                 const float* __restrict__ W,
                                                 const float* __restrict__ bias,
                                                 OutT* __restrict__ out) {
  __shared__ _Float16 lA[128 * LDA_P];
  __shared__ _Float16 lB[128 * LDA_P];
  const int tid  = threadIdx.x;
  const int m0   = blockIdx.x * 128;
  const int n0   = blockIdx.y * 128;
  const int lane = tid & 31, l15 = lane & 15, lh = lane >> 4;
  const int wave = tid >> 5, wy = wave & 1, wx = wave >> 1;

  v8f acc[4][2] = {};

  for (int k0 = 0; k0 < DDIM; k0 += 32) {
    stage_f32(lA, X + (size_t)m0 * DDIM + k0, DDIM, tid);
    stage_f32(lB, W + (size_t)n0 * DDIM + k0, DDIM, tid);
    __syncthreads();
    v16h af[4], bf[2];
#pragma unroll
    for (int mi = 0; mi < 4; ++mi)
      af[mi] = load_fragA(lA, wy * 64 + mi * 16 + l15, lh);
#pragma unroll
    for (int ni = 0; ni < 2; ++ni)
      bf[ni] = load_fragB(lB, wx * 32 + ni * 16 + l15, lh);
#pragma unroll
    for (int mi = 0; mi < 4; ++mi)
#pragma unroll
      for (int ni = 0; ni < 2; ++ni)
        acc[mi][ni] = __builtin_amdgcn_wmma_f32_16x16x32_f16(
            false, af[mi], false, bf[ni], (short)0, acc[mi][ni], false, false);
    __syncthreads();
  }

#pragma unroll
  for (int mi = 0; mi < 4; ++mi)
#pragma unroll
    for (int ni = 0; ni < 2; ++ni) {
      const int n = n0 + wx * 32 + ni * 16 + l15;
      const float bv = bias[n];
#pragma unroll
      for (int j = 0; j < 8; ++j) {
        const int m = m0 + wy * 64 + mi * 16 + j + 8 * lh;
        out[(size_t)m * DDIM + n] = (OutT)(acc[mi][ni][j] + bv);
      }
    }
}

// ---------------- score GEMM: score[b,lag] += sum_t q[t+lag].k[t] --------
// Per batch, lower-triangle 128x128 tiles of Q K^T; async global->LDS tile
// staging; per-tile lag binning in LDS (ds_add_f32), then 255 global float
// atomics per block.
__global__ __launch_bounds__(256) void score_gemm(const _Float16* __restrict__ Q,
                                                  const _Float16* __restrict__ Kmat,
                                                  float* __restrict__ score) {
  const int ti = blockIdx.x, tj = blockIdx.y, b = blockIdx.z;
  if (ti < tj) return;  // all lags negative in this tile
  __shared__ _Float16 lA[128 * LDA_P];
  __shared__ _Float16 lB[128 * LDA_P];
  __shared__ float bins[256];

  const int i0 = ti * 128, j0 = tj * 128;
  const _Float16* Qb = Q    + (size_t)b * TDIM * DDIM;
  const _Float16* Kb = Kmat + (size_t)b * TDIM * DDIM;
  const int tid  = threadIdx.x;
  const int lane = tid & 31, l15 = lane & 15, lh = lane >> 4;
  const int wave = tid >> 5, wy = wave & 1, wx = wave >> 1;

  v8f acc[4][2] = {};

  for (int k0 = 0; k0 < DDIM; k0 += 32) {
    stage_f16_async(lA, Qb + (size_t)i0 * DDIM + k0, DDIM, tid);
    stage_f16_async(lB, Kb + (size_t)j0 * DDIM + k0, DDIM, tid);
    wait_async_lds();
    __syncthreads();
    v16h af[4], bf[2];
#pragma unroll
    for (int mi = 0; mi < 4; ++mi)
      af[mi] = load_fragA(lA, wy * 64 + mi * 16 + l15, lh);
#pragma unroll
    for (int ni = 0; ni < 2; ++ni)
      bf[ni] = load_fragB(lB, wx * 32 + ni * 16 + l15, lh);
#pragma unroll
    for (int mi = 0; mi < 4; ++mi)
#pragma unroll
      for (int ni = 0; ni < 2; ++ni)
        acc[mi][ni] = __builtin_amdgcn_wmma_f32_16x16x32_f16(
            false, af[mi], false, bf[ni], (short)0, acc[mi][ni], false, false);
    __syncthreads();
  }

  bins[tid] = 0.0f;
  __syncthreads();

  const int base = i0 - j0;  // lag = base + (m - n), bin = (m - n) + 127
#pragma unroll
  for (int mi = 0; mi < 4; ++mi)
#pragma unroll
    for (int ni = 0; ni < 2; ++ni) {
      const int n = wx * 32 + ni * 16 + l15;
#pragma unroll
      for (int j = 0; j < 8; ++j) {
        const int m = wy * 64 + mi * 16 + j + 8 * lh;
        if (base + m - n >= 0)
          atomicAdd(&bins[m - n + 127], acc[mi][ni][j]);
      }
    }
  __syncthreads();

  if (tid < 255) {
    const int lag = base + tid - 127;
    if (lag >= 0) atomicAdd(&score[(size_t)b * TDIM + lag], bins[tid]);
  }
}

// ---------------- top-k selection (one block per batch) ----------------
__global__ __launch_bounds__(256) void topk_kernel(const float* __restrict__ score,
                                                   const int* __restrict__ topk_p,
                                                   int* __restrict__ lags) {
  __shared__ float s[TDIM];
  __shared__ float rv[256];
  __shared__ int   ri[256];
  const int b = blockIdx.x, tid = threadIdx.x;
  for (int i = tid; i < TDIM; i += 256) s[i] = score[(size_t)b * TDIM + i];
  __syncthreads();
  if (tid == 0) s[0] = -3.4e38f;  // mask zero lag
  __syncthreads();
  int K = *topk_p;
  if (K > 8) K = 8;
  if (K < 1) K = 1;
  for (int r = 0; r < K; ++r) {
    float best = -3.4e38f;
    int bi = 0;
    for (int i = tid; i < TDIM; i += 256) {
      const float v = s[i];
      if (v > best) { best = v; bi = i; }
    }
    rv[tid] = best; ri[tid] = bi;
    __syncthreads();
    for (int off = 128; off > 0; off >>= 1) {
      if (tid < off && rv[tid + off] > rv[tid]) {
        rv[tid] = rv[tid + off]; ri[tid] = ri[tid + off];
      }
      __syncthreads();
    }
    if (tid == 0) {
      lags[b * 8 + r] = ri[0];
      s[ri[0]] = -3.4e38f;
    }
    __syncthreads();
  }
}

// ---------------- gather: out = (1/K) * sum_i shift(v, lag_i) ----------------
__global__ __launch_bounds__(128) void gather_kernel(const float* __restrict__ v,
                                                     const int* __restrict__ lags,
                                                     const int* __restrict__ topk_p,
                                                     float* __restrict__ out) {
  const int blk = blockIdx.x;
  const int b = blk >> 12;       // / 4096
  const int t = blk & (TDIM - 1);
  int K = *topk_p;
  if (K > 8) K = 8;
  if (K < 1) K = 1;
  const int d = threadIdx.x;     // 0..127 float4 per row
  const float4* vb = (const float4*)v + (size_t)b * TDIM * (DDIM / 4);
  float4 acc; acc.x = acc.y = acc.z = acc.w = 0.0f;
  for (int i = 0; i < K; ++i) {
    const int lag = lags[b * 8 + i];
    const int src = (t - lag) & (TDIM - 1);
    const float4 x = vb[(size_t)src * (DDIM / 4) + d];
    acc.x += x.x; acc.y += x.y; acc.z += x.z; acc.w += x.w;
  }
  const float sc = 1.0f / (float)K;
  float4 r; r.x = acc.x * sc; r.y = acc.y * sc; r.z = acc.z * sc; r.w = acc.w * sc;
  ((float4*)out)[(size_t)blk * (DDIM / 4) + d] = r;
}

// ---------------- launch ----------------
extern "C" void kernel_launch(void* const* d_in, const int* in_sizes, int n_in,
                              void* d_out, int out_size, void* d_ws, size_t ws_size,
                              hipStream_t stream) {
  const float* x   = (const float*)d_in[0];
  const float* Wq  = (const float*)d_in[1];
  const float* bq  = (const float*)d_in[2];
  const float* Wk  = (const float*)d_in[3];
  const float* bk  = (const float*)d_in[4];
  const float* Wv  = (const float*)d_in[5];
  const float* bv  = (const float*)d_in[6];
  const int*   tkp = (const int*)d_in[7];

  char* ws = (char*)d_ws;
  const size_t QH = 0;                                   // f16 q: 32 MB
  const size_t KH = QH + (size_t)BDIM * TDIM * DDIM * 2; // f16 k: 32 MB
  const size_t VF = KH + (size_t)BDIM * TDIM * DDIM * 2; // f32 v: 64 MB
  const size_t SC = VF + (size_t)BDIM * TDIM * DDIM * 4; // f32 score
  const size_t LG = SC + (size_t)BDIM * TDIM * 4;        // int lags

  _Float16* qh    = (_Float16*)(ws + QH);
  _Float16* kh    = (_Float16*)(ws + KH);
  float*    vf    = (float*)(ws + VF);
  float*    score = (float*)(ws + SC);
  int*      lags  = (int*)(ws + LG);

  hipMemsetAsync(score, 0, (size_t)BDIM * TDIM * sizeof(float), stream);

  dim3 gp((BDIM * TDIM) / 128, DDIM / 128);
  proj_gemm<_Float16><<<gp, 256, 0, stream>>>(x, Wq, bq, qh);
  proj_gemm<_Float16><<<gp, 256, 0, stream>>>(x, Wk, bk, kh);
  proj_gemm<float><<<gp, 256, 0, stream>>>(x, Wv, bv, vf);

  dim3 gs(TDIM / 128, TDIM / 128, BDIM);
  score_gemm<<<gs, 256, 0, stream>>>(qh, kh, score);

  topk_kernel<<<BDIM, 256, 0, stream>>>(score, tkp, lags);

  gather_kernel<<<BDIM * TDIM, 128, 0, stream>>>(vf, lags, tkp, (float*)d_out);
}
